// GraphGPS_11716670783507
// MI455X (gfx1250) — compile-verified, matrix operand order checked by reading, not run on hardware
//
#include <hip/hip_runtime.h>

// ---------------- problem constants ----------------
#define GN   32768      // G*N nodes
#define DM   128        // model dim
#define DINC 64         // input dim
#define EC   262144     // edges
#define GC   256        // graphs
#define NC   128        // nodes per graph
#define HC   4          // heads
#define HDC  32         // head dim
#define LC   3          // layers

typedef __bf16 bf16_t;
typedef __attribute__((ext_vector_type(16))) __bf16 v16bf;
typedef __attribute__((ext_vector_type(8)))  float  v8f;

// 1/sqrt(1+1e-5) folded at compile time
#define BN_RSQRT 0.99999500003749969f

// ---------------- bf16 helpers (pure-integer RNE convert) ----------------
static __device__ __forceinline__ bf16_t f2bf(float f) {
    unsigned u = __builtin_bit_cast(unsigned, f);
    unsigned short h = (unsigned short)((u + 0x7FFFu + ((u >> 16) & 1u)) >> 16);
    return __builtin_bit_cast(bf16_t, h);
}
static __device__ __forceinline__ float bf2f(bf16_t x) {
    unsigned short s = __builtin_bit_cast(unsigned short, x);
    unsigned u = ((unsigned)s) << 16;
    return __builtin_bit_cast(float, u);
}

// ---------------- WMMA wrapper ----------------
static __device__ __forceinline__ v8f wmma_bf16(v16bf a, v16bf b, v8f c) {
    return __builtin_amdgcn_wmma_f32_16x16x32_bf16(
        false, a, false, b, (short)0, c, false, false);
}

// ---------------- fragment loaders ----------------
// A fragment (16x32 bf16): lanes 0-15 hold K {0..7,16..23}, lanes 16-31 K {8..15,24..31}
static __device__ __forceinline__ v16bf load_a_f32(const float* A, int lda,
                                                   int row0, int k0, int lane) {
    int r  = row0 + (lane & 15);
    int kb = (lane & 16) ? 8 : 0;
    const float* p = A + (long)r * lda + k0 + kb;
    v16bf o;
#pragma unroll
    for (int i = 0; i < 8; ++i) { o[i] = f2bf(p[i]); o[8 + i] = f2bf(p[16 + i]); }
    return o;
}
static __device__ __forceinline__ v16bf load_a_bf(const bf16_t* A, int lda,
                                                  int row0, int k0, int lane) {
    int r  = row0 + (lane & 15);
    int kb = (lane & 16) ? 8 : 0;
    const bf16_t* p = A + (long)r * lda + k0 + kb;
    v16bf o;
#pragma unroll
    for (int i = 0; i < 8; ++i) { o[i] = p[i]; o[8 + i] = p[16 + i]; }
    return o;
}
// B fragment (32x16): element B[k][n] at base[k*kStride + n*nStride]
static __device__ __forceinline__ v16bf load_b_bf(const bf16_t* B, int kStride,
                                                  int nStride, int k0, int col0,
                                                  int lane) {
    int c  = col0 + (lane & 15);
    int kb = (lane & 16) ? 8 : 0;
    const bf16_t* p = B + (long)(k0 + kb) * kStride + (long)c * nStride;
    v16bf o;
#pragma unroll
    for (int i = 0; i < 8; ++i) { o[i] = p[i * kStride]; o[8 + i] = p[(16 + i) * kStride]; }
    return o;
}

// ---------------- f32 -> bf16 weight conversion ----------------
__global__ void convert_bf16_kernel(const float* __restrict__ src,
                                    bf16_t* __restrict__ dst, int n) {
    int i = blockIdx.x * blockDim.x + threadIdx.x;
    for (; i < n; i += gridDim.x * blockDim.x) dst[i] = f2bf(src[i]);
}

// ---------------- WMMA GEMM, 16x64 strip per wave, compile-time epilogue ----
// C[M,N] = epi( A[M,K] @ W[K,N] + bias )
// FLAGS: 1=RELU, 2=+res, 4=BN(gamma,beta), 8=+add2 (order: bias,res,BN,add2,relu)
template <int FLAGS>
__global__ __launch_bounds__(256) void gemm_bf16_kernel(
    const float* __restrict__ A, const bf16_t* __restrict__ W,
    const float* __restrict__ bias, float* __restrict__ C,
    int M, int N, int K,
    const float* __restrict__ res, const float* __restrict__ gamma,
    const float* __restrict__ beta, const float* __restrict__ add2) {
    constexpr bool RELU = FLAGS & 1, RES = FLAGS & 2, BN = FLAGS & 4, ADD2 = FLAGS & 8;
    int lane   = threadIdx.x & 31;
    int wave   = threadIdx.x >> 5;
    int strips = N >> 6;                          // 64-wide strips
    int wid    = blockIdx.x * (blockDim.x >> 5) + wave;
    int tm     = wid / strips, ts = wid % strips;
    if (tm >= (M >> 4)) return;                   // wave-uniform guard
    int row0 = tm << 4, col0 = ts << 6;

    v8f acc[4] = {};
    for (int k0 = 0; k0 < K; k0 += 32) {
        v16bf a = load_a_f32(A, K, row0, k0, lane);   // shared across 4 N-tiles
#pragma unroll
        for (int j = 0; j < 4; ++j) {
            v16bf b = load_b_bf(W, N, 1, k0, col0 + j * 16, lane);
            acc[j]  = wmma_bf16(a, b, acc[j]);
        }
    }
    int rbase = row0 + ((lane >> 4) << 3);
#pragma unroll
    for (int j = 0; j < 4; ++j) {
        int   c    = col0 + j * 16 + (lane & 15);
        float bv   = bias[c];
        float bn_s = 0.0f, bn_b = 0.0f;
        if (BN) { bn_s = gamma[c] * BN_RSQRT; bn_b = beta[c]; }
#pragma unroll
        for (int r = 0; r < 8; ++r) {
            long idx = (long)(rbase + r) * N + c;
            float v  = acc[j][r] + bv;
            if (RES)  v += res[idx];
            if (BN)   v  = v * bn_s + bn_b;
            if (ADD2) v += add2[idx];
            if (RELU) v  = fmaxf(v, 0.0f);
            C[idx] = v;
        }
    }
}

// ---------------- z = h (vectorized copy) ----------------
__global__ void copy4_kernel(const float4* __restrict__ src,
                             float4* __restrict__ dst, int n4) {
    int i = blockIdx.x * blockDim.x + threadIdx.x;
    if (i < n4) dst[i] = src[i];
}

// ---------------- edge message + scatter-add (ea folded, rank-1) ----------
__global__ void edge_msg_kernel(const float* __restrict__ h,
                                const int* __restrict__ ei,
                                const float* __restrict__ eattr,
                                const float* __restrict__ ew,
                                const float* __restrict__ eb,
                                float* __restrict__ z) {
    long gid = (long)blockIdx.x * blockDim.x + threadIdx.x;
    int  e   = (int)(gid >> 5);
    if (e >= EC) return;
    int d0  = (int)(gid & 31) << 2;
    int src = ei[e];
    int dst = ei[EC + e];
    float a = eattr[e];
    const float4 hv = *(const float4*)(h + (long)src * DM + d0);
    const float4 wv = *(const float4*)(ew + d0);
    const float4 bv = *(const float4*)(eb + d0);
    float*       zd = z + (long)dst * DM + d0;
    atomicAdd(&zd[0], fmaxf(hv.x + a * wv.x + bv.x, 0.0f));
    atomicAdd(&zd[1], fmaxf(hv.y + a * wv.y + bv.y, 0.0f));
    atomicAdd(&zd[2], fmaxf(hv.z + a * wv.z + bv.z, 0.0f));
    atomicAdd(&zd[3], fmaxf(hv.w + a * wv.w + bv.w, 0.0f));
}

// ---------------- fused per-(graph,head) attention ----------------
__global__ __launch_bounds__(128) void attn_kernel(
    const float* __restrict__ h, const bf16_t* __restrict__ Wqkv /*[128,384]*/,
    const float* __restrict__ bqkv /*[384]*/, float* __restrict__ o /*[GN,128]*/) {
    int g = blockIdx.x >> 2, head = blockIdx.x & 3;
    int tid = threadIdx.x, lane = tid & 31, wave = tid >> 5;
    int base = g * NC;

    __shared__ bf16_t sHS[NC * DM];        // h_g, then S/P (32 KB)
    __shared__ bf16_t sQ[NC * HDC];        // 8 KB each
    __shared__ bf16_t sK[NC * HDC];
    __shared__ bf16_t sV[NC * HDC];

    for (int i = tid; i < NC * DM; i += 128) {
        int r = i >> 7, cd = i & 127;
        sHS[i] = f2bf(h[(long)(base + r) * DM + cd]);
    }
    __syncthreads();

    // Q/K/V = h_g @ Wqkv[:, sel] + b   (K = 128 -> 4 WMMA steps)
    for (int which = 0; which < 3; ++which) {
        int colW0 = which * DM + head * HDC;
        bf16_t* dst = (which == 0) ? sQ : ((which == 1) ? sK : sV);
        for (int t = wave; t < 16; t += 4) {
            int tm = t >> 1, tn = t & 1;
            v8f acc = {};
#pragma unroll
            for (int k0 = 0; k0 < DM; k0 += 32) {
                v16bf a = load_a_bf(sHS, DM, tm * 16, k0, lane);
                v16bf b = load_b_bf(Wqkv, 3 * DM, 1, k0, colW0 + tn * 16, lane);
                acc = wmma_bf16(a, b, acc);
            }
            int cc = tn * 16 + (lane & 15);
            int rb = tm * 16 + ((lane >> 4) << 3);
            float bv = bqkv[colW0 + cc];
#pragma unroll
            for (int r = 0; r < 8; ++r) dst[(rb + r) * HDC + cc] = f2bf(acc[r] + bv);
        }
    }
    __syncthreads();

    // S = (Q K^T) * 1/sqrt(HD)  -> sHS  (single WMMA per tile, K = 32)
    const float scale = 0.17677669529663687f;
    for (int t = wave; t < 64; t += 4) {
        int tm = t >> 3, tn = t & 7;
        v16bf a = load_a_bf(sQ, HDC, tm * 16, 0, lane);
        v16bf b = load_b_bf(sK, 1, HDC, 0, tn * 16, lane);  // B[k][j] = K[j][k]
        v8f acc = {};
        acc = wmma_bf16(a, b, acc);
        int cc = tn * 16 + (lane & 15);
        int rb = tm * 16 + ((lane >> 4) << 3);
#pragma unroll
        for (int r = 0; r < 8; ++r) sHS[(rb + r) * NC + cc] = f2bf(acc[r] * scale);
    }
    __syncthreads();

    // softmax per row (thread r handles row r)
    {
        bf16_t* row = sHS + tid * NC;
        float mx = -3.4e38f;
        for (int j = 0; j < NC; ++j) mx = fmaxf(mx, bf2f(row[j]));
        float s = 0.0f;
        for (int j = 0; j < NC; ++j) {
            float e = __expf(bf2f(row[j]) - mx);
            s += e;
            row[j] = f2bf(e);
        }
        float inv = 1.0f / s;
        for (int j = 0; j < NC; ++j) row[j] = f2bf(bf2f(row[j]) * inv);
    }
    __syncthreads();

    // O = P @ V   (K = 128 -> 4 WMMA steps)
    for (int t = wave; t < 16; t += 4) {
        int tm = t >> 1, tn = t & 1;
        v8f acc = {};
#pragma unroll
        for (int k0 = 0; k0 < NC; k0 += 32) {
            v16bf a = load_a_bf(sHS, NC, tm * 16, k0, lane);
            v16bf b = load_b_bf(sV, HDC, 1, k0, tn * 16, lane);
            acc = wmma_bf16(a, b, acc);
        }
        int cc = tn * 16 + (lane & 15);
        int rb = tm * 16 + ((lane >> 4) << 3);
#pragma unroll
        for (int r = 0; r < 8; ++r)
            o[(long)(base + rb + r) * DM + head * HDC + cc] = acc[r];
    }
}

// ---------------- host-side launch helpers ----------------
static void launch_gemm(hipStream_t s, int flags, const float* A, const bf16_t* W,
                        const float* bias, float* C, int M, int N, int K,
                        const float* res, const float* gamma, const float* beta,
                        const float* add2) {
    int strips = (M >> 4) * (N >> 6);
    int blocks = strips / 8;
    switch (flags) {
    case 0:
        gemm_bf16_kernel<0><<<blocks, 256, 0, s>>>(A, W, bias, C, M, N, K, res, gamma, beta, add2);
        break;
    case 1:
        gemm_bf16_kernel<1><<<blocks, 256, 0, s>>>(A, W, bias, C, M, N, K, res, gamma, beta, add2);
        break;
    case 6:
        gemm_bf16_kernel<6><<<blocks, 256, 0, s>>>(A, W, bias, C, M, N, K, res, gamma, beta, add2);
        break;
    case 14:
        gemm_bf16_kernel<14><<<blocks, 256, 0, s>>>(A, W, bias, C, M, N, K, res, gamma, beta, add2);
        break;
    }
}

extern "C" void kernel_launch(void* const* d_in, const int* in_sizes, int n_in,
                              void* d_out, int out_size, void* d_ws, size_t ws_size,
                              hipStream_t stream) {
    (void)in_sizes; (void)n_in; (void)out_size; (void)ws_size;
    const float* x      = (const float*)d_in[0];   // [GN, 64]
    const int*   ei     = (const int*)d_in[1];     // [2, E]
    const float* eattr  = (const float*)d_in[2];   // [E]
    /* batch d_in[3] unused */
    const float* node_w = (const float*)d_in[4];   // [64,128]
    const float* node_b = (const float*)d_in[5];   // [128]
    const float* edge_w = (const float*)d_in[6];   // [1,128]
    const float* edge_b = (const float*)d_in[7];   // [128]
    const float* gw1 = (const float*)d_in[8];      // [3,128,128]
    const float* gb1 = (const float*)d_in[9];      // [3,128]
    const float* gw2 = (const float*)d_in[10];
    const float* gb2 = (const float*)d_in[11];
    const float* aiw = (const float*)d_in[12];     // [3,128,384]
    const float* aib = (const float*)d_in[13];     // [3,384]
    const float* aow = (const float*)d_in[14];
    const float* aob = (const float*)d_in[15];
    const float* mw1 = (const float*)d_in[16];     // [3,128,256]
    const float* mb1 = (const float*)d_in[17];     // [3,256]
    const float* mw2 = (const float*)d_in[18];     // [3,256,128]
    const float* mb2 = (const float*)d_in[19];
    const float* n1g = (const float*)d_in[20];
    const float* n1b = (const float*)d_in[21];
    const float* n2g = (const float*)d_in[22];
    const float* n2b = (const float*)d_in[23];
    const float* n3g = (const float*)d_in[24];
    const float* n3b = (const float*)d_in[25];

    // workspace layout
    const long ND = (long)GN * DM;                 // 4,194,304 floats
    float* zbuf = (float*)d_ws;                    // 16 MB
    float* t1   = zbuf + ND;                       // 16 MB (also attention O)
    float* h1b  = t1 + ND;                         // 16 MB (h1, then out in-place)
    float* hbuf = h1b + ND;                        // 16 MB (current h)
    float* f1   = zbuf;                            // 32 MB alias (z + t1 region)
    bf16_t* wa  = (bf16_t*)(hbuf + ND);            // bf16 weight area (~1 MB)

    bf16_t* nwb = wa;  wa += DINC * DM;            // node_w
    bf16_t *gw1b[LC], *gw2b[LC], *aiwb[LC], *aowb[LC], *mw1b[LC], *mw2b[LC];
    for (int l = 0; l < LC; ++l) {
        gw1b[l] = wa; wa += DM * DM;
        gw2b[l] = wa; wa += DM * DM;
        aiwb[l] = wa; wa += DM * 3 * DM;
        aowb[l] = wa; wa += DM * DM;
        mw1b[l] = wa; wa += DM * 2 * DM;
        mw2b[l] = wa; wa += 2 * DM * DM;
    }

    auto conv = [&](const float* s, bf16_t* d, int n) {
        convert_bf16_kernel<<<(n + 255) / 256, 256, 0, stream>>>(s, d, n);
    };
    conv(node_w, nwb, DINC * DM);
    for (int l = 0; l < LC; ++l) {
        conv(gw1 + (long)l * DM * DM,     gw1b[l], DM * DM);
        conv(gw2 + (long)l * DM * DM,     gw2b[l], DM * DM);
        conv(aiw + (long)l * DM * 3 * DM, aiwb[l], DM * 3 * DM);
        conv(aow + (long)l * DM * DM,     aowb[l], DM * DM);
        conv(mw1 + (long)l * DM * 2 * DM, mw1b[l], DM * 2 * DM);
        conv(mw2 + (long)l * 2 * DM * DM, mw2b[l], 2 * DM * DM);
    }

    // node encoder: h = x @ node_w + node_b
    launch_gemm(stream, 0, x, nwb, node_b, hbuf, GN, DM, DINC,
                nullptr, nullptr, nullptr, nullptr);

    for (int l = 0; l < LC; ++l) {
        const float* lb1 = gb1 + (long)l * DM;
        const float* lb2 = gb2 + (long)l * DM;
        const float* lab = aib + (long)l * 3 * DM;
        const float* lob = aob + (long)l * DM;
        const float* lm1 = mb1 + (long)l * 2 * DM;
        const float* lm2 = mb2 + (long)l * DM;

        // 1-2. z = h + scatter_add(relu(h[src] + ea))
        copy4_kernel<<<(int)(ND / 4 / 256), 256, 0, stream>>>(
            (const float4*)hbuf, (float4*)zbuf, (int)(ND / 4));
        edge_msg_kernel<<<(EC * 32) / 256, 256, 0, stream>>>(
            hbuf, ei, eattr, edge_w, edge_b, zbuf);

        // 3. t1 = relu(z @ W1 + b1)
        launch_gemm(stream, 1, zbuf, gw1b[l], lb1, t1, GN, DM, DM,
                    nullptr, nullptr, nullptr, nullptr);
        // 4. h1 = BN(t1 @ W2 + b2 + h)
        launch_gemm(stream, 6, t1, gw2b[l], lb2, h1b, GN, DM, DM,
                    hbuf, n1g + (long)l * DM, n1b + (long)l * DM, nullptr);

        // 5. attention: o (=t1) from h
        attn_kernel<<<GC * HC, 128, 0, stream>>>(hbuf, aiwb[l], lab, t1);

        // 6. out = h1 + BN(o @ Wout + b + h)   (in-place over h1b)
        launch_gemm(stream, 14, t1, aowb[l], lob, h1b, GN, DM, DM,
                    hbuf, n2g + (long)l * DM, n2b + (long)l * DM, h1b);

        // 7. f1 = relu(out @ mlp_w1 + b1)   [GN, 256]
        launch_gemm(stream, 1, h1b, mw1b[l], lm1, f1, GN, 2 * DM, DM,
                    nullptr, nullptr, nullptr, nullptr);
        // 8. h_next = BN(out + f1 @ mlp_w2 + b2)
        float* dest = (l == LC - 1) ? (float*)d_out : hbuf;
        launch_gemm(stream, 6, f1, mw2b[l], lm2, dest, GN, DM, 2 * DM,
                    h1b, n3g + (long)l * DM, n3b + (long)l * DM, nullptr);
    }
}
